// SQLinear_85684597555513
// MI455X (gfx1250) — 8-bit matrix operand order repaired and checked by reading, compiled, not run on hardware
//
#include <hip/hip_runtime.h>
#include <stdint.h>

typedef __attribute__((ext_vector_type(8))) int v8i;

#define S_DIM   2048
#define B_DIM   8
#define IN_DIM  1024
#define OUT_DIM 4096
#define M_DIM   (S_DIM * B_DIM)
#define QMAXF   127.0f


__global__ void sq_init(unsigned* __restrict__ act_u, unsigned* __restrict__ w_u) {
    for (int i = threadIdx.x; i < IN_DIM; i += 256) {
        act_u[i] = 0u;
        w_u[i]   = 0u;
    }
}

__global__ void sq_colmax(const float* __restrict__ src, unsigned* __restrict__ dst,
                          int rows_per_block) {
    const int col = blockIdx.x * 256 + threadIdx.x;
    const long r0 = (long)blockIdx.y * rows_per_block;
    const float* p = src + r0 * IN_DIM + col;
    float m = 0.0f;
    for (int r = 0; r < rows_per_block; ++r)
        m = fmaxf(m, fabsf(p[(long)r * IN_DIM]));
    atomicMax(dst + col, __float_as_uint(m));
}

__global__ void sq_scale(const unsigned* __restrict__ act_u, const unsigned* __restrict__ w_u,
                         float* __restrict__ scale, float* __restrict__ inv_scale,
                         float* __restrict__ svals) {
    __shared__ float redx[256];
    __shared__ float redw[256];
    const int t = threadIdx.x;
    float mx = 0.0f, mw = 0.0f;
    for (int j = 0; j < IN_DIM / 256; ++j) {
        const int k = t + j * 256;
        const float a = __uint_as_float(act_u[k]);
        const float w = __uint_as_float(w_u[k]);
        float s = sqrtf(a) / sqrtf(w);
        if (s == 0.0f) s = 1.0f;
        const float is = 1.0f / s;
        scale[k]     = s;
        inv_scale[k] = is;
        mx = fmaxf(mx, a * is);
        mw = fmaxf(mw, w * s);
    }
    redx[t] = mx; redw[t] = mw;
    __syncthreads();
    for (int off = 128; off > 0; off >>= 1) {
        if (t < off) {
            redx[t] = fmaxf(redx[t], redx[t + off]);
            redw[t] = fmaxf(redw[t], redw[t + off]);
        }
        __syncthreads();
    }
    if (t == 0) {
        const float ax = redx[0], aw = redw[0];
        const float sx = (ax > 0.0f) ? ax / QMAXF : 1.0f;
        const float sw = (aw > 0.0f) ? aw / QMAXF : 1.0f;
        svals[0] = 1.0f / sx;
        svals[1] = 1.0f / sw;
        svals[2] = sx * sw;
    }
}

__device__ __forceinline__ int sq_q8(float v, float c, float f) {
    float q = rintf(v * c * f);
    q = fminf(QMAXF, fmaxf(-QMAXF, q));
    return (int)q;
}

__global__ void sq_quant(const float* __restrict__ src, const float* __restrict__ colmul,
                         const float* __restrict__ svals, int sidx,
                         unsigned* __restrict__ dst, long n4) {
    const long g = (long)blockIdx.x * blockDim.x + threadIdx.x;
    if (g >= n4) return;
    const float f = svals[sidx];
    const long  e = g * 4;
    const int   k = (int)(e & (IN_DIM - 1));
    const float4 v  = *(const float4*)(src + e);
    const float4 cm = *(const float4*)(colmul + k);
    const int b0 = sq_q8(v.x, cm.x, f);
    const int b1 = sq_q8(v.y, cm.y, f);
    const int b2 = sq_q8(v.z, cm.z, f);
    const int b3 = sq_q8(v.w, cm.w, f);
    dst[g] = (unsigned)(b0 & 0xFF) | ((unsigned)(b1 & 0xFF) << 8) |
             ((unsigned)(b2 & 0xFF) << 16) | ((unsigned)(b3 & 0xFF) << 24);
}

__device__ __forceinline__ v8i sq_load_a(const signed char* p) {
    const int2 c0 = *(const int2*)(p +  0);
    const int2 c1 = *(const int2*)(p + 16);
    const int2 c2 = *(const int2*)(p + 32);
    const int2 c3 = *(const int2*)(p + 48);
    v8i r = {c0.x, c0.y, c1.x, c1.y, c2.x, c2.y, c3.x, c3.y};
    return r;
}
__device__ __forceinline__ v8i sq_load_b(const signed char* p) {
    const int2 c0 = *(const int2*)(p +  0);
    const int2 c1 = *(const int2*)(p + 16);
    const int2 c2 = *(const int2*)(p + 32);
    const int2 c3 = *(const int2*)(p + 48);
    v8i r = {c0.x, c0.y, c1.x, c1.y, c2.x, c2.y, c3.x, c3.y};
    return r;
}

__global__ __launch_bounds__(256)
void sq_gemm(const signed char* __restrict__ qx, const signed char* __restrict__ qw,
             const float* __restrict__ bias, const float* __restrict__ svals,
             float* __restrict__ out) {
    const int lane = threadIdx.x & 31;
    const int wave = threadIdx.x >> 5;
    const int lo   = lane & 15;
    const int hi   = lane >> 4;

    const int n0 = blockIdx.x * 32;
    const int m0 = (blockIdx.y * 8 + wave) * 32;

    const signed char* a0 = qx + (size_t)(m0 + lo) * IN_DIM + hi * 8;
    const signed char* a1 = a0 + (size_t)16 * IN_DIM;
    const signed char* b0 = qw + (size_t)(n0 + lo) * IN_DIM + hi * 8;
    const signed char* b1 = b0 + (size_t)16 * IN_DIM;

    v8i acc00 = {0, 0, 0, 0, 0, 0, 0, 0};
    v8i acc01 = {0, 0, 0, 0, 0, 0, 0, 0};
    v8i acc10 = {0, 0, 0, 0, 0, 0, 0, 0};
    v8i acc11 = {0, 0, 0, 0, 0, 0, 0, 0};

    for (int kb = 0; kb < IN_DIM; kb += 64) {
        const v8i A0 = sq_load_a(a0 + kb);
        const v8i A1 = sq_load_a(a1 + kb);
        const v8i B0 = sq_load_b(b0 + kb);
        const v8i B1 = sq_load_b(b1 + kb);
        acc00 = __builtin_amdgcn_wmma_i32_16x16x64_iu8(true, A0, true, B0, acc00, false, false);
        acc01 = __builtin_amdgcn_wmma_i32_16x16x64_iu8(true, A0, true, B1, acc01, false, false);
        acc10 = __builtin_amdgcn_wmma_i32_16x16x64_iu8(true, A1, true, B0, acc10, false, false);
        acc11 = __builtin_amdgcn_wmma_i32_16x16x64_iu8(true, A1, true, B1, acc11, false, false);
    }

    const float sprod = svals[2];
    const float bz0 = bias[n0 + lo];
    const float bz1 = bias[n0 + 16 + lo];

    float* o0 = out + (size_t)(m0 + hi * 8) * OUT_DIM + n0 + lo;
    float* o1 = o0 + (size_t)16 * OUT_DIM;
#pragma unroll
    for (int j = 0; j < 8; ++j) {
        const size_t ro = (size_t)j * OUT_DIM;
        o0[ro]      = (float)acc00[j] * sprod + bz0;
        o0[ro + 16] = (float)acc01[j] * sprod + bz1;
        o1[ro]      = (float)acc10[j] * sprod + bz0;
        o1[ro + 16] = (float)acc11[j] * sprod + bz1;
    }
}

extern "C" void kernel_launch(void* const* d_in, const int* in_sizes, int n_in,
                              void* d_out, int out_size, void* d_ws, size_t ws_size,
                              hipStream_t stream) {
    (void)in_sizes; (void)n_in; (void)out_size; (void)ws_size;

    const float* x    = (const float*)d_in[0];
    const float* w    = (const float*)d_in[1];
    const float* bias = (const float*)d_in[2];
    float* out = (float*)d_out;

    char* ws = (char*)d_ws;
    signed char* qx = (signed char*)ws;
    signed char* qw = (signed char*)(ws + (size_t)M_DIM * IN_DIM);
    char* p = ws + (size_t)M_DIM * IN_DIM + (size_t)OUT_DIM * IN_DIM;
    unsigned* act_u     = (unsigned*)p;  p += IN_DIM * sizeof(unsigned);
    unsigned* w_u       = (unsigned*)p;  p += IN_DIM * sizeof(unsigned);
    float*    scale     = (float*)p;     p += IN_DIM * sizeof(float);
    float*    inv_scale = (float*)p;     p += IN_DIM * sizeof(float);
    float*    svals     = (float*)p;

    sq_init<<<1, 256, 0, stream>>>(act_u, w_u);

    sq_colmax<<<dim3(IN_DIM / 256, M_DIM / 256),   256, 0, stream>>>(x, act_u, 256);
    sq_colmax<<<dim3(IN_DIM / 256, OUT_DIM / 256), 256, 0, stream>>>(w, w_u,   256);

    sq_scale<<<1, 256, 0, stream>>>(act_u, w_u, scale, inv_scale, svals);

    const long n4x = (long)M_DIM * IN_DIM / 4;
    sq_quant<<<(int)((n4x + 255) / 256), 256, 0, stream>>>(x, inv_scale, svals, 0,
                                                           (unsigned*)qx, n4x);
    const long n4w = (long)OUT_DIM * IN_DIM / 4;
    sq_quant<<<(int)((n4w + 255) / 256), 256, 0, stream>>>(w, scale, svals, 1,
                                                           (unsigned*)qw, n4w);

    sq_gemm<<<dim3(OUT_DIM / 32, M_DIM / 256), 256, 0, stream>>>(qx, qw, bias, svals, out);
}
